// TripletLoss_21328807592557
// MI455X (gfx1250) — matrix operand order repaired and checked by reading, not compiled, not run on hardware
//
#include <hip/hip_runtime.h>
#include <hip/hip_bf16.h>

typedef __attribute__((ext_vector_type(16))) _Float16 v16h;
typedef __attribute__((ext_vector_type(8)))  _Float16 v8h;
typedef __attribute__((ext_vector_type(4)))  _Float16 v4h;
typedef __attribute__((ext_vector_type(8)))  float    v8f;
typedef __attribute__((ext_vector_type(4)))  float    v4f;

constexpr int   BATCH  = 4096;
constexpr int   DIM    = 1024;
constexpr float MARGIN = 1.0f;
constexpr float BIG    = 1e9f;

// ---------------------------------------------------------------------------
// Phase 1: L2-normalize rows of both fp32 matrices, emit f16 rows into d_ws.
// One 256-thread block per row (8 waves, wave32). 4 floats / thread.
// ---------------------------------------------------------------------------
__global__ __launch_bounds__(256) void normalize_f16_kernel(
    const float* __restrict__ text, const float* __restrict__ graph,
    _Float16* __restrict__ t_h, _Float16* __restrict__ g_h)
{
    __shared__ float red[256];
    const int row = blockIdx.x;
    const int tid = threadIdx.x;

    const float*  src;
    _Float16*     dst;
    if (row < BATCH) { src = text  + (size_t)row * DIM;           dst = t_h + (size_t)row * DIM; }
    else             { src = graph + (size_t)(row - BATCH) * DIM; dst = g_h + (size_t)(row - BATCH) * DIM; }

    v4f v = *(const v4f*)(src + tid * 4);
    red[tid] = v[0]*v[0] + v[1]*v[1] + v[2]*v[2] + v[3]*v[3];
    __syncthreads();
    #pragma unroll
    for (int s = 128; s > 0; s >>= 1) {
        if (tid < s) red[tid] += red[tid + s];
        __syncthreads();
    }
    const float scale = 1.0f / fmaxf(sqrtf(red[0]), 1e-12f);

    v4h h;
    #pragma unroll
    for (int i = 0; i < 4; ++i) h[i] = (_Float16)(v[i] * scale);
    *(v4h*)(dst + tid * 4) = h;
}

// ---------------------------------------------------------------------------
// Phase 2: distances = 1 - t_h . g_h^T via V_WMMA_F32_16X16X32_F16.
// Each wave computes a 32(M) x 64(N) register tile: two A fragments and four
// B fragments per k-step feed 8 WMMAs -> 1.5 b128 loads per WMMA (operands
// are fully L2-resident at 16 MB total, so this is the L2-traffic knob).
// Grid: 128 M-tiles x 64 N-supertiles = 8192 waves = 1024 blocks of 8 waves.
// ---------------------------------------------------------------------------
__global__ __launch_bounds__(256) void gemm_dist_kernel(
    const _Float16* __restrict__ t_h, const _Float16* __restrict__ g_h,
    float* __restrict__ dist)
{
    const int lane   = threadIdx.x & 31;
    const int wave   = threadIdx.x >> 5;
    const int wgid   = blockIdx.x * 8 + wave;     // 0..8191
    const int mTile  = wgid & 127;                // 0..127 (32 rows each)
    const int nSuper = wgid >> 7;                 // 0..63  (64 cols each)
    const int mBase  = mTile * 32;
    const int nBase  = nSuper * 64;

    // A fragment addressing (16x32 f16): lane<16 -> K {0..7,16..23}, else {8..15,24..31}
    const int am  = lane & 15;
    const int ak0 = (lane < 16) ? 0 : 8;
    const _Float16* aRow0 = t_h + (size_t)(mBase + am) * DIM + ak0;        // M sub-tile 0
    const _Float16* aRow1 = aRow0 + (size_t)16 * DIM;                      // M sub-tile 1

    // B fragment addressing (32x16 f16): column = lane&15 (a row of g), lane<16 -> K {0..7,16..23}, else {8..15,24..31}
    const int bc  = lane & 15;
    const int bk0 = (lane < 16) ? 0 : 8;
    const _Float16* bPtr[4];
    #pragma unroll
    for (int t = 0; t < 4; ++t)
        bPtr[t] = g_h + (size_t)(nBase + 16 * t + bc) * DIM + bk0;

    union AFrag { v16h v; struct { v8h lo, hi; } p; };

    v8f acc0[4] = {v8f{}, v8f{}, v8f{}, v8f{}};   // rows mBase..mBase+15
    v8f acc1[4] = {v8f{}, v8f{}, v8f{}, v8f{}};   // rows mBase+16..mBase+31

    for (int kb = 0; kb < DIM; kb += 32) {
        AFrag a0, a1;
        a0.p.lo = *(const v8h*)(aRow0 + kb);
        a0.p.hi = *(const v8h*)(aRow0 + kb + 16);
        a1.p.lo = *(const v8h*)(aRow1 + kb);
        a1.p.hi = *(const v8h*)(aRow1 + kb + 16);
        #pragma unroll
        for (int t = 0; t < 4; ++t) {
            AFrag b;
            b.p.lo = *(const v8h*)(bPtr[t] + kb);
            b.p.hi = *(const v8h*)(bPtr[t] + kb + 16);
            acc0[t] = __builtin_amdgcn_wmma_f32_16x16x32_f16(
                false, a0.v, false, b.v, (short)0, acc0[t], false, false);
            acc1[t] = __builtin_amdgcn_wmma_f32_16x16x32_f16(
                false, a1.v, false, b.v, (short)0, acc1[t], false, false);
        }
    }

    // C/D layout: lanes 0-15: VGPR r -> M=r; lanes 16-31: VGPR r -> M=8+r. N = lane&15.
    const int col   = nBase + (lane & 15);
    const int rOff  = (lane < 16) ? 0 : 8;
    #pragma unroll
    for (int t = 0; t < 4; ++t) {
        #pragma unroll
        for (int r = 0; r < 8; ++r) {
            dist[(size_t)(mBase +      rOff + r) * BATCH + col + 16 * t] = 1.0f - acc0[t][r];
            dist[(size_t)(mBase + 16 + rOff + r) * BATCH + col + 16 * t] = 1.0f - acc1[t][r];
        }
    }
}

// ---------------------------------------------------------------------------
// Phase 3a: zero the loss accumulator (harness poisons d_out).
// ---------------------------------------------------------------------------
__global__ void zero_loss_kernel(float* loss)
{
    if (threadIdx.x == 0 && blockIdx.x == 0) *loss = 0.0f;
}

// ---------------------------------------------------------------------------
// Phase 3b: per-row hard-negative mining + loss accumulation.
// One 256-thread block per row; LDS min/any reduction; atomicAdd of per_row/B.
// ---------------------------------------------------------------------------
__global__ __launch_bounds__(256) void mine_kernel(
    const float* __restrict__ dist, const int* __restrict__ labels,
    float* __restrict__ loss)
{
    __shared__ float smin[256];
    __shared__ int   sneg[256];
    const int i   = blockIdx.x;
    const int tid = threadIdx.x;
    const int li  = labels[i];
    const float* row = dist + (size_t)i * BATCH;

    float mn  = BIG;
    int   any = 0;
    for (int j = tid; j < BATCH; j += 256) {
        float d = row[j];
        if (labels[j] != li) { any = 1; mn = fminf(mn, d); }
    }
    smin[tid] = mn; sneg[tid] = any;
    __syncthreads();
    #pragma unroll
    for (int s = 128; s > 0; s >>= 1) {
        if (tid < s) {
            smin[tid] = fminf(smin[tid], smin[tid + s]);
            sneg[tid] |= sneg[tid + s];
        }
        __syncthreads();
    }
    if (tid == 0) {
        float per_row = 0.0f;
        if (sneg[0]) per_row = fmaxf(row[i] - smin[0] + MARGIN, 0.0f);
        atomicAdd(loss, per_row * (1.0f / (float)BATCH));
    }
}

// ---------------------------------------------------------------------------
extern "C" void kernel_launch(void* const* d_in, const int* in_sizes, int n_in,
                              void* d_out, int out_size, void* d_ws, size_t ws_size,
                              hipStream_t stream)
{
    (void)in_sizes; (void)n_in; (void)out_size; (void)ws_size;
    const float* text   = (const float*)d_in[0];
    const float* graph  = (const float*)d_in[1];
    const int*   labels = (const int*)d_in[2];

    float* out  = (float*)d_out;
    float* dist = out + 1;                       // outputs: [loss, distances(B*B)]

    _Float16* t_h = (_Float16*)d_ws;             // 8 MB
    _Float16* g_h = t_h + (size_t)BATCH * DIM;   // 8 MB

    normalize_f16_kernel<<<2 * BATCH, 256, 0, stream>>>(text, graph, t_h, g_h);
    gemm_dist_kernel   <<<1024,      256, 0, stream>>>(t_h, g_h, dist);
    zero_loss_kernel   <<<1,           1, 0, stream>>>(out);
    mine_kernel        <<<BATCH,     256, 0, stream>>>(dist, labels, out);
}
